// ANet_37580963840019
// MI455X (gfx1250) — matrix operand order repaired and checked by reading, not compiled, not run on hardware
//
#include <hip/hip_runtime.h>
#include <hip/hip_bf16.h>
#include <stddef.h>

#define NN 1024
#define EE 2048
#define ET 3072   // E + N self loops
#define HH 32

typedef _Float16 h16;
typedef __attribute__((ext_vector_type(16))) _Float16 v16h;
typedef __attribute__((ext_vector_type(8)))  _Float16 v8h;
typedef __attribute__((ext_vector_type(8)))  float    v8f;

// ---------------- edge list with self loops ----------------
__global__ void k_edges(const int* __restrict__ ei, int* __restrict__ sdx, int* __restrict__ ddx) {
    int e = blockIdx.x * blockDim.x + threadIdx.x;
    if (e >= ET) return;
    if (e < EE) { sdx[e] = ei[e]; ddx[e] = ei[EE + e]; }
    else        { sdx[e] = e - EE; ddx[e] = e - EE; }
}

// ---------------- weight f32[K,Nout] -> f16 transposed [Nout,Kpad] ----------------
__global__ void k_wT(const float* __restrict__ w, h16* __restrict__ wT,
                     int K, int Kpad, int Nout) {
    long long i = (long long)blockIdx.x * blockDim.x + threadIdx.x;
    long long total = (long long)Nout * Kpad;
    if (i >= total) return;
    int k = (int)(i % Kpad);
    long long n = i / Kpad;
    wT[i] = (k < K) ? (h16)w[(long long)k * Nout + n] : (h16)0.0f;
}

// ---------------- activation f32[M,K] -> f16[M,Kpad], fused LeakyReLU ----------------
__global__ void k_act16(const float* __restrict__ x, h16* __restrict__ a,
                        int K, int Kpad, float slope, int M) {
    int i = blockIdx.x * blockDim.x + threadIdx.x;
    if (i >= M * Kpad) return;
    int k = i % Kpad, m = i / Kpad;
    float v = 0.0f;
    if (k < K) { v = x[m * K + k]; v = v > 0.0f ? v : slope * v; }
    a[i] = (h16)v;
}

// ---------------- WMMA GEMM: C[M,Nout] = A[M,Kpad](f16) @ B(=WT[Nout,Kpad]) + bias ----------------
// one wave -> 16x64 tile (4 x v_wmma_f32_16x16x32_f16 per K-step), 8 waves / block
__global__ void __launch_bounds__(256)
k_gemm(const h16* __restrict__ A, const h16* __restrict__ BT,
       const float* __restrict__ bias, float* __restrict__ C,
       int M, int Kpad, int Nout) {
    const int lane = threadIdx.x & 31;
    const int wave = threadIdx.x >> 5;
    const int n0 = (blockIdx.x * 8 + wave) * 64;
    if (n0 >= Nout) return;
    const int m0  = blockIdx.y * 16;
    const int sel = lane >> 4;          // lane group 0 / 1
    const int l15 = lane & 15;

    const h16* arow = A + (size_t)(m0 + l15) * Kpad + (sel << 3);
    const h16* bcol = BT + (size_t)(n0 + l15) * Kpad + (sel << 3);

    v8f z8 = {};
    v8f acc[4];
#pragma unroll
    for (int j = 0; j < 4; ++j) acc[j] = z8;

    for (int k0 = 0; k0 < Kpad; k0 += 32) {
        v8h lo = *(const v8h*)(arow + k0);
        v8h hi = *(const v8h*)(arow + k0 + 16);
        v16h a;
#pragma unroll
        for (int i = 0; i < 8; ++i) { a[i] = lo[i]; a[i + 8] = hi[i]; }
#pragma unroll
        for (int j = 0; j < 4; ++j) {
            v8h blo = *(const v8h*)(bcol + (size_t)j * 16 * Kpad + k0);
            v8h bhi = *(const v8h*)(bcol + (size_t)j * 16 * Kpad + k0 + 16);
            v16h b;
#pragma unroll
            for (int i = 0; i < 8; ++i) { b[i] = blo[i]; b[i + 8] = bhi[i]; }
            acc[j] = __builtin_amdgcn_wmma_f32_16x16x32_f16(
                false, a, false, b, (short)0, acc[j], false, false);
        }
    }
    // store: lane -> column n, VGPR i -> row m0 + sel*8 + i
#pragma unroll
    for (int j = 0; j < 4; ++j) {
        int n = n0 + j * 16 + l15;
        float bv = bias[n];
#pragma unroll
        for (int i = 0; i < 8; ++i)
            C[(size_t)(m0 + sel * 8 + i) * Nout + n] = acc[j][i] + bv;
    }
}

// ---------------- attention scores: one wave per (edge, head) ----------------
__global__ void k_score(const float* __restrict__ xl, const float* __restrict__ xr,
                        const float* __restrict__ att, const int* __restrict__ sdx,
                        const int* __restrict__ ddx, float* __restrict__ score,
                        int C, int D) {
    int gw = (blockIdx.x * blockDim.x + threadIdx.x) >> 5;
    int lane = threadIdx.x & 31;
    if (gw >= ET * HH) return;
    int e = gw / HH, h = gw % HH;
    const float* pl = xl + (size_t)sdx[e] * D + h * C;
    const float* pr = xr + (size_t)ddx[e] * D + h * C;
    const float* pa = att + h * C;
    float acc = 0.0f;
    for (int c = lane; c < C; c += 32) {
        float v = pl[c] + pr[c];
        v = v > 0.0f ? v : 0.2f * v;
        acc += v * pa[c];
    }
    for (int o = 16; o >= 1; o >>= 1) acc += __shfl_xor(acc, o);
    if (lane == 0) score[e * HH + h] = acc;
}

__device__ __forceinline__ unsigned f_enc(float f) {
    unsigned u = __float_as_uint(f);
    return (u & 0x80000000u) ? ~u : (u | 0x80000000u);
}
__device__ __forceinline__ float f_dec(unsigned u) {
    return (u & 0x80000000u) ? __uint_as_float(u ^ 0x80000000u) : __uint_as_float(~u);
}

__global__ void k_init(unsigned* __restrict__ smax, float* __restrict__ denom) {
    int i = blockIdx.x * blockDim.x + threadIdx.x;
    if (i < NN * HH) { smax[i] = 0u; denom[i] = 0.0f; }
}

__global__ void k_smax(const float* __restrict__ score, const int* __restrict__ ddx,
                       unsigned* __restrict__ smax) {
    int i = blockIdx.x * blockDim.x + threadIdx.x;
    if (i >= ET * HH) return;
    int e = i / HH, h = i % HH;
    atomicMax(&smax[ddx[e] * HH + h], f_enc(score[i]));
}

__global__ void k_alpha(float* __restrict__ score, const int* __restrict__ ddx,
                        const unsigned* __restrict__ smax, float* __restrict__ denom) {
    int i = blockIdx.x * blockDim.x + threadIdx.x;
    if (i >= ET * HH) return;
    int e = i / HH, h = i % HH;
    float a = expf(score[i] - f_dec(smax[ddx[e] * HH + h]));
    score[i] = a;
    atomicAdd(&denom[ddx[e] * HH + h], a);
}

__global__ void k_zero(float* __restrict__ p, long long n) {
    long long i = (long long)blockIdx.x * blockDim.x + threadIdx.x;
    if (i < n) p[i] = 0.0f;
}

// out[dst] += (alpha/denom) * xl[src]; block.y = edge, block.x over D
__global__ void k_aggr(const float* __restrict__ xl, const float* __restrict__ alpha,
                       const float* __restrict__ denom, const int* __restrict__ sdx,
                       const int* __restrict__ ddx, float* __restrict__ agg,
                       int C, int D) {
    int e = blockIdx.y;
    int idx = blockIdx.x * blockDim.x + threadIdx.x;
    if (idx >= D) return;
    int h = idx / C;
    int s = sdx[e], d = ddx[e];
    float w = alpha[e * HH + h] / denom[d * HH + h];
    atomicAdd(&agg[(size_t)d * D + idx], w * xl[(size_t)s * D + idx]);
}

// head mean + bias
__global__ void k_mean(const float* __restrict__ agg, const float* __restrict__ gb,
                       float* __restrict__ z, int C, int D) {
    int i = blockIdx.x * blockDim.x + threadIdx.x;
    if (i >= NN * C) return;
    int n = i / C, c = i % C;
    const float* p = agg + (size_t)n * D + c;
    float s = 0.0f;
#pragma unroll 1
    for (int h = 0; h < HH; ++h) s += p[h * C];
    z[i] = s * (1.0f / HH) + gb[c];
}

// final 128 -> 2 layer (tiny; plain), fused input LeakyReLU(0.01)
__global__ void k_dec3(const float* __restrict__ g2, const float* __restrict__ w,
                       const float* __restrict__ b, float* __restrict__ out) {
    int i = blockIdx.x * blockDim.x + threadIdx.x;
    if (i >= NN * 2) return;
    int n = i >> 1, j = i & 1;
    float s = b[j];
    for (int k = 0; k < 128; ++k) {
        float v = g2[n * 128 + k];
        v = v > 0.0f ? v : 0.01f * v;
        s += v * w[k * 2 + j];
    }
    out[i] = s;
}

extern "C" void kernel_launch(void* const* d_in, const int* in_sizes, int n_in,
                              void* d_out, int out_size, void* d_ws, size_t ws_size,
                              hipStream_t stream) {
    (void)in_sizes; (void)n_in; (void)out_size; (void)ws_size;
    const float* x   = (const float*)d_in[0];
    const int*   ei  = (const int*)d_in[1];
    const float* w1l = (const float*)d_in[2];  const float* b1l = (const float*)d_in[3];
    const float* w1r = (const float*)d_in[4];  const float* b1r = (const float*)d_in[5];
    const float* att1= (const float*)d_in[6];  const float* gb1 = (const float*)d_in[7];
    const float* w2l = (const float*)d_in[8];  const float* b2l = (const float*)d_in[9];
    const float* w2r = (const float*)d_in[10]; const float* b2r = (const float*)d_in[11];
    const float* att2= (const float*)d_in[12]; const float* gb2 = (const float*)d_in[13];
    const float* w3l = (const float*)d_in[14]; const float* b3l = (const float*)d_in[15];
    const float* w3r = (const float*)d_in[16]; const float* b3r = (const float*)d_in[17];
    const float* att3= (const float*)d_in[18]; const float* gb3 = (const float*)d_in[19];
    const float* dw1 = (const float*)d_in[20]; const float* db1 = (const float*)d_in[21];
    const float* dw2 = (const float*)d_in[22]; const float* db2 = (const float*)d_in[23];
    const float* dw3 = (const float*)d_in[24]; const float* db3 = (const float*)d_in[25];
    float* out = (float*)d_out;
    char* ws = (char*)d_ws;

    size_t off = 0;
    auto carve = [&](size_t bytes) -> char* {
        char* p = ws + off;
        off = (off + bytes + 255) & ~(size_t)255;
        return p;
    };
    int*      sdx   = (int*)carve(ET * sizeof(int));
    int*      ddx   = (int*)carve(ET * sizeof(int));
    h16*      w1lT  = (h16*)carve((size_t)4096  * 32   * 2);
    h16*      w1rT  = (h16*)carve((size_t)4096  * 32   * 2);
    h16*      w2lT  = (h16*)carve((size_t)16384 * 128  * 2);
    h16*      w2rT  = (h16*)carve((size_t)16384 * 128  * 2);
    h16*      w3lT  = (h16*)carve((size_t)32896 * 512  * 2);
    h16*      w3rT  = (h16*)carve((size_t)32896 * 512  * 2);
    h16*      dw1T  = (h16*)carve((size_t)512   * 1056 * 2);
    h16*      dw2T  = (h16*)carve((size_t)128   * 512  * 2);
    h16*      a16   = (h16*)carve((size_t)NN * 1056 * 2);
    float*    xl    = (float*)carve((size_t)NN * 32896 * 4);
    float*    xr    = (float*)carve((size_t)NN * 32896 * 4);   // reused as agg
    float*    sco   = (float*)carve((size_t)ET * HH * 4);
    unsigned* smax  = (unsigned*)carve((size_t)NN * HH * 4);
    float*    denom = (float*)carve((size_t)NN * HH * 4);
    float*    zA    = (float*)carve((size_t)NN * 1028 * 4);
    float*    zB    = (float*)carve((size_t)NN * 1028 * 4);

    k_edges<<<(ET + 255) / 256, 256, 0, stream>>>(ei, sdx, ddx);

    auto convW = [&](const float* w, h16* wT, int K, int Kpad, int Nout) {
        long long total = (long long)Nout * Kpad;
        k_wT<<<(int)((total + 255) / 256), 256, 0, stream>>>(w, wT, K, Kpad, Nout);
    };
    convW(w1l, w1lT, 4, 32, 4096);     convW(w1r, w1rT, 4, 32, 4096);
    convW(w2l, w2lT, 128, 128, 16384); convW(w2r, w2rT, 128, 128, 16384);
    convW(w3l, w3lT, 512, 512, 32896); convW(w3r, w3rT, 512, 512, 32896);
    convW(dw1, dw1T, 1028, 1056, 512); convW(dw2, dw2T, 512, 512, 128);

    auto gat = [&](const float* zin, int cin, int kpad, float slope,
                   const h16* wlT, const float* bl, const h16* wrT, const float* br,
                   const float* att, const float* gb, int C, float* zout) {
        int D = HH * C;
        k_act16<<<(NN * kpad + 255) / 256, 256, 0, stream>>>(zin, a16, cin, kpad, slope, NN);
        dim3 gg((D / 64 + 7) / 8, NN / 16);
        k_gemm<<<gg, 256, 0, stream>>>(a16, wlT, bl, xl, NN, kpad, D);
        k_gemm<<<gg, 256, 0, stream>>>(a16, wrT, br, xr, NN, kpad, D);
        k_score<<<(ET * HH * 32 + 255) / 256, 256, 0, stream>>>(xl, xr, att, sdx, ddx, sco, C, D);
        k_init<<<(NN * HH + 255) / 256, 256, 0, stream>>>(smax, denom);
        k_smax<<<(ET * HH + 255) / 256, 256, 0, stream>>>(sco, ddx, smax);
        k_alpha<<<(ET * HH + 255) / 256, 256, 0, stream>>>(sco, ddx, smax, denom);
        long long nag = (long long)NN * D;
        k_zero<<<(int)((nag + 255) / 256), 256, 0, stream>>>(xr, nag);
        dim3 ga((D + 255) / 256, ET);
        k_aggr<<<ga, 256, 0, stream>>>(xl, sco, denom, sdx, ddx, xr, C, D);
        k_mean<<<(NN * C + 255) / 256, 256, 0, stream>>>(xr, gb, zout, C, D);
    };

    gat(x,   4,   32,  1.00f, w1lT, b1l, w1rT, b1r, att1, gb1, 128,  zA);
    gat(zA,  128, 128, 0.01f, w2lT, b2l, w2rT, b2r, att2, gb2, 512,  zB);
    gat(zB,  512, 512, 0.01f, w3lT, b3l, w3rT, b3r, att3, gb3, 1028, zA);

    // decoder: 1028 -> 512 -> 128 -> 2
    k_act16<<<(NN * 1056 + 255) / 256, 256, 0, stream>>>(zA, a16, 1028, 1056, 1.0f, NN);
    {
        dim3 gg((512 / 64 + 7) / 8, NN / 16);
        k_gemm<<<gg, 256, 0, stream>>>(a16, dw1T, db1, zB, NN, 1056, 512);
    }
    k_act16<<<(NN * 512 + 255) / 256, 256, 0, stream>>>(zB, a16, 512, 512, 0.01f, NN);
    {
        dim3 gg((128 / 64 + 7) / 8, NN / 16);
        k_gemm<<<gg, 256, 0, stream>>>(a16, dw2T, db2, zA, NN, 512, 128);
    }
    k_dec3<<<(NN * 2 + 255) / 256, 256, 0, stream>>>(zA, dw3, db3, out);
}
